// MonetaBlock_26920855011941
// MI455X (gfx1250) — compile-verified, matrix operand order checked by reading, not run on hardware
//
#include <hip/hip_runtime.h>
#include <math.h>

// ---------------- problem constants (match reference setup_inputs) ----------
constexpr int B_   = 4;
constexpr int NSEQ = 4096;
constexpr int DIM  = 256;
constexpr int D3   = 3 * DIM;          // 768
constexpr int MROW = B_ * NSEQ;        // 16384 rows for the GEMMs

typedef float     v2f  __attribute__((ext_vector_type(2)));
typedef float     v8f  __attribute__((ext_vector_type(8)));
typedef _Float16  v16h __attribute__((ext_vector_type(16)));

__device__ __forceinline__ float sigmoidf_(float z) {
    return 1.0f / (1.0f + __expf(-z));
}

// ======================================================================
// WMMA GEMM:  C[M,N] = X[M,K] * W[N,K]^T   (optionally * sigmoid(gatePre))
// fp32 path: V_WMMA_F32_16X16X4_F32, K-step 4.
// fallback:  V_WMMA_F32_16X16X32_F16 (convert on the fly), K-step 32.
// One wave computes one 16x16 output tile; 4 waves / block.
// ======================================================================
__global__ __launch_bounds__(128) void wmma_gemm_kernel(
    const float* __restrict__ X, const float* __restrict__ W,
    float* __restrict__ C, const float* __restrict__ gatePre,
    int M, int N, int K)
{
    const int wave = threadIdx.x >> 5;
    const int lane = threadIdx.x & 31;
    const int tileN = blockIdx.x * 4 + wave;
    const int tileM = blockIdx.y;
    if (tileN * 16 >= N || tileM * 16 >= M) return;

    const int row0 = tileM * 16;
    const int col0 = tileN * 16;
    const int mn   = lane & 15;     // M index for A-frag, N index for B-frag
    const int half = lane >> 4;     // K-half selector

    v8f acc = {};

#if __has_builtin(__builtin_amdgcn_wmma_f32_16x16x4_f32)
    // A 16x4 f32: VGPR0 = K{0|2}, VGPR1 = K{1|3}; lanes<16 -> K0/1, lanes>=16 -> K2/3
    for (int k = 0; k < K; k += 4) {
        v2f a, b;
        const float* xr = X + (size_t)(row0 + mn) * K + k + 2 * half;
        a.x = xr[0]; a.y = xr[1];
        const float* wr = W + (size_t)(col0 + mn) * K + k + 2 * half;
        b.x = wr[0]; b.y = wr[1];
        acc = __builtin_amdgcn_wmma_f32_16x16x4_f32(
            false, a, false, b, (short)0, acc, false, false);
    }
#else
    for (int k = 0; k < K; k += 32) {
        v16h a, b;
        const float* xr = X + (size_t)(row0 + mn) * K + k;
        const float* wr = W + (size_t)(col0 + mn) * K + k;
        #pragma unroll
        for (int j = 0; j < 8; ++j) {
            int kb = (j < 4) ? (half * 8 + 2 * j)
                             : (16 + half * 8 + 2 * (j - 4));
            a[2 * j]     = (_Float16)xr[kb];
            a[2 * j + 1] = (_Float16)xr[kb + 1];
            b[2 * j]     = (_Float16)wr[kb];
            b[2 * j + 1] = (_Float16)wr[kb + 1];
        }
        acc = __builtin_amdgcn_wmma_f32_16x16x32_f16(
            false, a, false, b, (short)0, acc, false, false);
    }
#endif

    // C/D layout: VGPR v, lanes<16 -> M=v, lanes>=16 -> M=v+8; N = lane&15
    const int nCol  = col0 + mn;
    const int mBase = row0 + (half ? 8 : 0);
    #pragma unroll
    for (int v = 0; v < 8; ++v) {
        size_t off = (size_t)(mBase + v) * N + nCol;
        float val = acc[v];
        if (gatePre) val *= sigmoidf_(gatePre[off]);
        C[off] = val;
    }
}

// ======================================================================
// Prep: depthwise causal conv(4) + RoPE + l2norm on q,k; conv on v;
//       eta/alpha = sigmoid(x . pg_w + pg_b).   One block per (b,n).
// ======================================================================
__device__ __forceinline__ float block_reduce256(float v, float* red) {
    int t = threadIdx.x;
    red[t] = v; __syncthreads();
    #pragma unroll
    for (int s = 128; s > 0; s >>= 1) {
        if (t < s) red[t] += red[t + s];
        __syncthreads();
    }
    float r = red[0];
    __syncthreads();
    return r;
}

__global__ __launch_bounds__(256) void prep_kernel(
    const float* __restrict__ x,     const float* __restrict__ qkv,
    const float* __restrict__ fcos,  const float* __restrict__ fsin,
    const float* __restrict__ qcw,   const float* __restrict__ qcb,
    const float* __restrict__ kcw,   const float* __restrict__ kcb,
    const float* __restrict__ vcw,   const float* __restrict__ vcb,
    const float* __restrict__ pgw,   const float* __restrict__ pgb,
    float* __restrict__ qh, float* __restrict__ kh, float* __restrict__ vh,
    float* __restrict__ eta, float* __restrict__ alpha)
{
    const int bn = blockIdx.x;
    const int b  = bn / NSEQ;
    const int n  = bn % NSEQ;
    const int d  = threadIdx.x;

    __shared__ float sQ[DIM], sK[DIM];
    __shared__ float red[256];

    // depthwise causal conv, kernel width 4 (taps at n-3..n)
    float accq = qcb[d], acck = kcb[d], accv = vcb[d];
    #pragma unroll
    for (int j = 0; j < 4; ++j) {
        int nn = n - 3 + j;
        if (nn >= 0) {
            const float* row = qkv + ((size_t)b * NSEQ + nn) * D3;
            accq += row[d]           * qcw[d * 4 + j];
            acck += row[DIM + d]     * kcw[d * 4 + j];
            accv += row[2 * DIM + d] * vcw[d * 4 + j];
        }
    }
    sQ[d] = accq; sK[d] = acck;
    __syncthreads();

    // RoPE
    const int p = d >> 1;
    const float c = fcos[(size_t)n * (DIM / 2) + p];
    const float s = fsin[(size_t)n * (DIM / 2) + p];
    const float q0 = sQ[d & ~1], q1 = sQ[d | 1];
    const float k0 = sK[d & ~1], k1 = sK[d | 1];
    const float qr = (d & 1) ? (q0 * s + q1 * c) : (q0 * c - q1 * s);
    const float kr = (d & 1) ? (k0 * s + k1 * c) : (k0 * c - k1 * s);

    // l2 norms
    const float qs = block_reduce256(qr * qr, red);
    const float ks = block_reduce256(kr * kr, red);
    const float qinv = 1.0f / fmaxf(sqrtf(qs), 1e-12f);
    const float kinv = 1.0f / fmaxf(sqrtf(ks), 1e-12f);

    // eta / alpha gates
    const size_t base = ((size_t)b * NSEQ + n) * DIM;
    const float xv = x[base + d];
    const float p0 = block_reduce256(xv * pgw[d], red);
    const float p1 = block_reduce256(xv * pgw[DIM + d], red);
    if (d == 0) {
        eta  [bn] = sigmoidf_(p0 + pgb[0]);
        alpha[bn] = sigmoidf_(p1 + pgb[1]);
    }

    qh[base + d] = qr * qinv;
    kh[base + d] = kr * kinv;
    vh[base + d] = accv;
}

// ======================================================================
// Sequential scan: one block per batch, 1024 threads.
// Each thread owns an 8x8 register tile of the 256x256 state A.
// W_t = A_t / denom_t is kept implicitly as (A in regs, scale in LDS).
// ======================================================================
__global__ __launch_bounds__(1024) void scan_kernel(
    const float* __restrict__ qh, const float* __restrict__ kh,
    const float* __restrict__ vh, const float* __restrict__ eta,
    const float* __restrict__ alpha, float* __restrict__ y)
{
    const int b   = blockIdx.x;
    const int tid = threadIdx.x;
    const int cg  = tid & 31;       // column group 0..31
    const int rgp = tid >> 5;       // row group    0..31
    const int j0  = cg * 8;
    const int i0  = rgp * 8;

    __shared__ float sK[DIM], sV[DIM], sQ[DIM], sG[DIM];
    __shared__ float part[32 * DIM];      // matvec partials (32 KB)
    __shared__ float red[1024 + 32];
    __shared__ float sScale;              // 1/denom of previous step

    float A[8][8];
    #pragma unroll
    for (int r = 0; r < 8; ++r)
        #pragma unroll
        for (int c = 0; c < 8; ++c) A[r][c] = 0.0f;

    if (tid == 0) sScale = 1.0f;

    for (int n = 0; n < NSEQ; ++n) {
        const size_t base = ((size_t)b * NSEQ + n) * DIM;
        if (tid < DIM) {
            sK[tid] = kh[base + tid];
            sV[tid] = vh[base + tid];
            sQ[tid] = qh[base + tid];
            if (n + 1 < NSEQ) {   // prefetch next timestep (global_prefetch_b8)
                __builtin_prefetch(kh + base + DIM + tid, 0, 1);
                __builtin_prefetch(qh + base + DIM + tid, 0, 1);
                __builtin_prefetch(vh + base + DIM + tid, 0, 1);
            }
        }
        if (tid == 1024 - 1) {
            red[1024] = eta  [b * NSEQ + n];
            red[1025] = alpha[b * NSEQ + n];
        }
        __syncthreads();

        float kreg[8], qreg[8];
        #pragma unroll
        for (int r = 0; r < 8; ++r) { kreg[r] = sK[i0 + r]; qreg[r] = sQ[i0 + r]; }

        // ---- pass 1: pred_j = scale * sum_i k_i A_ij  (partials per row group)
        #pragma unroll
        for (int c = 0; c < 8; ++c) {
            float p = 0.0f;
            #pragma unroll
            for (int r = 0; r < 8; ++r) p += kreg[r] * A[r][c];
            part[rgp * DIM + j0 + c] = p;
        }
        __syncthreads();

        if (tid < DIM) {
            float pr = 0.0f;
            for (int g = 0; g < 32; ++g) pr += part[g * DIM + tid];
            pr = (n == 0) ? sK[tid]            // W0 = I  ->  pred = k
                          : pr * sScale;       // W = A_prev / denom_prev
            const float diff = pr - sV[tid];
            sG[tid] = 3.0f * tanhf(10.0f * diff) * diff * diff;
        }
        __syncthreads();

        // ---- pass 2 (fused): A = alpha*A - eta*k g^T ; A^4 partial ; y partial
        float greg[8];
        #pragma unroll
        for (int c = 0; c < 8; ++c) greg[c] = sG[j0 + c];
        const float et = red[1024], al = red[1025];
        float ekr[8];
        #pragma unroll
        for (int r = 0; r < 8; ++r) ekr[r] = et * kreg[r];

        float a4p = 0.0f;
        float yp[8];
        #pragma unroll
        for (int c = 0; c < 8; ++c) yp[c] = 0.0f;

        #pragma unroll
        for (int r = 0; r < 8; ++r) {
            #pragma unroll
            for (int c = 0; c < 8; ++c) {
                float a = al * A[r][c] - ekr[r] * greg[c];
                A[r][c] = a;
                float a2 = a * a;
                a4p += a2 * a2;
                yp[c] += qreg[r] * a;
            }
        }
        #pragma unroll
        for (int c = 0; c < 8; ++c) part[rgp * DIM + j0 + c] = yp[c];
        red[tid] = a4p;
        __syncthreads();

        // ---- reduce A^4 -> denom, publish scale
        if (tid < 256) {
            red[tid] = red[tid] + red[tid + 256] + red[tid + 512] + red[tid + 768];
        }
        __syncthreads();
        if (tid < 32) {
            float s8 = 0.0f;
            #pragma unroll
            for (int g = 0; g < 8; ++g) s8 += red[tid * 8 + g];
            red[1024 + tid] = s8;   // safe: eta/alpha already consumed
        }
        __syncthreads();
        if (tid == 0) {
            float tot = 0.0f;
            #pragma unroll
            for (int g = 0; g < 32; ++g) tot += red[1024 + g];
            sScale = 1.0f / (sqrtf(tot) + 1e-6f);
        }
        __syncthreads();

        // ---- y_j = scale * sum_i q_i A_ij
        if (tid < DIM) {
            float yv = 0.0f;
            for (int g = 0; g < 32; ++g) yv += part[g * DIM + tid];
            y[base + tid] = yv * sScale;
        }
        __syncthreads();   // sScale carries to next step's pred
    }
}

// ======================================================================
extern "C" void kernel_launch(void* const* d_in, const int* in_sizes, int n_in,
                              void* d_out, int out_size, void* d_ws, size_t ws_size,
                              hipStream_t stream)
{
    const float* x      = (const float*)d_in[0];
    const float* fcos   = (const float*)d_in[1];
    const float* fsin   = (const float*)d_in[2];
    const float* qkv_w  = (const float*)d_in[3];
    const float* qcw    = (const float*)d_in[4];
    const float* qcb    = (const float*)d_in[5];
    const float* kcw    = (const float*)d_in[6];
    const float* kcb    = (const float*)d_in[7];
    const float* vcw    = (const float*)d_in[8];
    const float* vcb    = (const float*)d_in[9];
    const float* pgw    = (const float*)d_in[10];
    const float* pgb    = (const float*)d_in[11];
    // d_in[12] = W0 (identity) handled analytically in scan step 0
    const float* gate_w = (const float*)d_in[13];
    const float* out_w  = (const float*)d_in[14];
    float* out = (float*)d_out;

    // workspace layout (floats). qkv region is reused for gate-pre and y
    // after prep_kernel has consumed qkv.
    float* ws    = (float*)d_ws;
    float* qkv   = ws;                                   // 16384*768
    float* gpre  = ws;                                   // overlays qkv[0 .. 4M)
    float* yBuf  = ws + (size_t)MROW * DIM;              // overlays qkv[4M .. 8M)
    float* qh    = ws + (size_t)MROW * D3;               // 16384*256
    float* kh    = qh + (size_t)MROW * DIM;
    float* vh    = kh + (size_t)MROW * DIM;
    float* etaB  = vh + (size_t)MROW * DIM;              // 16384
    float* alpB  = etaB + MROW;                          // 16384
    (void)ws_size; (void)n_in; (void)in_sizes; (void)out_size;

    // 1) qkv = x @ qkv_w^T            (WMMA)
    wmma_gemm_kernel<<<dim3(D3 / 64, MROW / 16), 128, 0, stream>>>(
        x, qkv_w, qkv, nullptr, MROW, D3, DIM);

    // 2) conv + rope + l2norm + eta/alpha
    prep_kernel<<<MROW, 256, 0, stream>>>(
        x, qkv, fcos, fsin, qcw, qcb, kcw, kcb, vcw, vcb, pgw, pgb,
        qh, kh, vh, etaB, alpB);

    // 3) gate pre-activation = x @ gate_w^T   (WMMA; overwrites qkv region)
    wmma_gemm_kernel<<<dim3(DIM / 64, MROW / 16), 128, 0, stream>>>(
        x, gate_w, gpre, nullptr, MROW, DIM, DIM);

    // 4) sequential Moneta scan (state in VGPRs, one WGP per batch)
    scan_kernel<<<B_, 1024, 0, stream>>>(qh, kh, vh, etaB, alpB, yBuf);

    // 5) out = (y @ out_w^T) * sigmoid(gpre)  (WMMA + fused gate epilogue)
    wmma_gemm_kernel<<<dim3(DIM / 64, MROW / 16), 128, 0, stream>>>(
        yBuf, out_w, out, gpre, MROW, DIM, DIM);
}
